// LocalAttention_15977278341424
// MI455X (gfx1250) — compile-verified, matrix operand order checked by reading, not run on hardware
//
#include <hip/hip_runtime.h>
#include <stdint.h>
#include <stddef.h>

// Problem constants (match reference)
#define B_    16
#define S_    4096
#define E_    512
#define H_    8
#define WIN_  64
#define DH_   64          // E/H
#define NTOK  (B_*S_)     // 65536 rows
#define NQKV  (3*E_)      // 1536

typedef __bf16 bf16_t;
typedef __attribute__((ext_vector_type(16))) __bf16 v16bf;
typedef __attribute__((ext_vector_type(8)))  float  v8f;

union Frag { v16bf v; uint4 q[2]; };

__device__ __forceinline__ unsigned short f2bf_raw(float f) {
  unsigned u = __builtin_bit_cast(unsigned, f);
  unsigned r = u + 0x7FFFu + ((u >> 16) & 1u);   // round-to-nearest-even
  return (unsigned short)(r >> 16);
}
__device__ __forceinline__ bf16_t f2bf(float f) {
  unsigned short h = f2bf_raw(f);
  return __builtin_bit_cast(bf16_t, h);
}
__device__ __forceinline__ unsigned pack2bf(float lo, float hi) {
  return ((unsigned)f2bf_raw(hi) << 16) | (unsigned)f2bf_raw(lo);
}

// ---------------------------------------------------------------------------
// f32 -> bf16 conversion (weights only; x conversion is fused into GEMM1)
// ---------------------------------------------------------------------------
__global__ __launch_bounds__(256) void cvt_f32_bf16_kernel(
    const float* __restrict__ src, bf16_t* __restrict__ dst, int n) {
  for (int i = blockIdx.x * blockDim.x + threadIdx.x; i < n;
       i += gridDim.x * blockDim.x)
    dst[i] = f2bf(src[i]);
}

// ---------------------------------------------------------------------------
// GEMM: C[M,N] = A[M,K] * W[N,K]^T + bias[N]
// Workgroup tile 64(M) x 512(N), 8 waves, wave tile 64x64 (16 accumulators).
// K-step 32 -> v_wmma_f32_16x16x32_bf16. A tile staged (and, for the QKV
// projection, converted f32->bf16 on the fly) in LDS.
// ---------------------------------------------------------------------------
template <bool A_F32, bool OUT_BF16>
__global__ __launch_bounds__(256) void gemm_wmma_kernel(
    const void*   __restrict__ Ain,  // M x K (f32 or bf16 row-major)
    const bf16_t* __restrict__ W,    // N x K row-major
    const float*  __restrict__ bias, // N
    void* __restrict__ out,          // M x N (bf16 or f32)
    int M, int N, int K) {
  __shared__ bf16_t As[64 * 32];     // 4 KB staged A tile (bf16)

  const int tid   = threadIdx.x;
  const int wave  = tid >> 5;
  const int lane  = tid & 31;
  const int lrow  = lane & 15;
  const int hi    = lane >> 4;
  const int khalf = hi << 3;                 // 0 or 8 (ISA 16-bit A/B layout)
  const int m0    = blockIdx.x * 64;
  const int nt    = blockIdx.y * 512 + wave * 64;

  v8f acc[4][4];
#pragma unroll
  for (int mt = 0; mt < 4; ++mt)
#pragma unroll
    for (int j = 0; j < 4; ++j) acc[mt][j] = (v8f){0,0,0,0,0,0,0,0};

  for (int kk = 0; kk < K; kk += 32) {
    __syncthreads();
    {   // cooperative stage of 64x32 A tile: 8 elements per thread
      int row = tid >> 2, ch = tid & 3;
      uint4 pack;
      if (A_F32) {
        const float* ap = (const float*)Ain + (size_t)(m0 + row) * K + kk + ch * 8;
        float4 f0 = *(const float4*)(ap);
        float4 f1 = *(const float4*)(ap + 4);
        pack.x = pack2bf(f0.x, f0.y);
        pack.y = pack2bf(f0.z, f0.w);
        pack.z = pack2bf(f1.x, f1.y);
        pack.w = pack2bf(f1.z, f1.w);
      } else {
        pack = *(const uint4*)((const bf16_t*)Ain + (size_t)(m0 + row) * K + kk + ch * 8);
      }
      ((uint4*)As)[tid] = pack;
    }
    __syncthreads();

    // A fragments persist across the 4 N sub-tiles
    Frag a[4];
#pragma unroll
    for (int mt = 0; mt < 4; ++mt) {
      const bf16_t* ap = As + (mt * 16 + lrow) * 32 + khalf;
      a[mt].q[0] = *(const uint4*)(ap);
      a[mt].q[1] = *(const uint4*)(ap + 16);
    }

#pragma unroll
    for (int j = 0; j < 4; ++j) {
      const bf16_t* wp = W + (size_t)(nt + j * 16 + lrow) * K + kk + khalf;
      __builtin_prefetch(wp + 32, 0, 0);   // global_prefetch_b8: next K-step of W
      Frag b;
      b.q[0] = *(const uint4*)(wp);
      b.q[1] = *(const uint4*)(wp + 16);
#pragma unroll
      for (int mt = 0; mt < 4; ++mt)
        acc[mt][j] = __builtin_amdgcn_wmma_f32_16x16x32_bf16(
            false, a[mt].v, false, b.v, (short)0, acc[mt][j], false, false);
    }
  }

  // Epilogue: C layout -> lane holds col N = lane&15, rows r + 8*(lane>>4)
#pragma unroll
  for (int mt = 0; mt < 4; ++mt)
#pragma unroll
    for (int j = 0; j < 4; ++j) {
      int n = nt + j * 16 + lrow;
      float bs = bias[n];
#pragma unroll
      for (int r = 0; r < 8; ++r) {
        int m = m0 + mt * 16 + r + 8 * hi;
        float vv = acc[mt][j][r] + bs;
        if (OUT_BF16) ((bf16_t*)out)[(size_t)m * N + n] = f2bf(vv);
        else          ((float*) out)[(size_t)m * N + n] = vv;
      }
    }
}

// ---------------------------------------------------------------------------
// Attention: one block per sequence position s, one wave per head.
// The attention axis is the batch dim (L=16), Dh=64.
//   S = (Q K^T)/8 ; P = softmax_N(S) ; O = P V
// Softmax is done by round-tripping the 16x16 tile through LDS: each of 16
// lanes owns one row in-lane (no cross-lane bpermute trees), and the same
// LDS pass re-shapes P from the WMMA C-layout into the A-fragment layout.
// ---------------------------------------------------------------------------
__global__ __launch_bounds__(256) void attn_wmma_kernel(
    const bf16_t* __restrict__ qkv,   // NTOK x 1536 (q|k|v)
    bf16_t* __restrict__ o) {         // NTOK x 512
  __shared__ float  Sf[H_][16 * 16];  // raw scores, 8 KB
  __shared__ bf16_t Pl[H_][16 * 16];  // softmaxed P, 4 KB
  __shared__ bf16_t vT[H_][DH_ * 16]; // per-head V^T (Dh x 16 keys), 16 KB

  const int s     = blockIdx.x;       // s = n*WIN + w
  const int h     = threadIdx.x >> 5;
  const int lane  = threadIdx.x & 31;
  const int lrow  = lane & 15;
  const int hi    = lane >> 4;
  const int khalf = hi << 3;

  const bf16_t* qbase = qkv + ((size_t)lrow * S_ + s) * NQKV + h * DH_;

  // ---- scores: 16x16 f32, K=64 over head dim (2 WMMAs) ----
  v8f sc = (v8f){0,0,0,0,0,0,0,0};
#pragma unroll
  for (int kf = 0; kf < 2; ++kf) {
    Frag qa, kb;
    const bf16_t* qp = qbase + kf * 32 + khalf;
    const bf16_t* kp = qp + E_;                 // k block of packed qkv
    qa.q[0] = *(const uint4*)(qp); qa.q[1] = *(const uint4*)(qp + 16);
    kb.q[0] = *(const uint4*)(kp); kb.q[1] = *(const uint4*)(kp + 16);
    sc = __builtin_amdgcn_wmma_f32_16x16x32_bf16(
        false, qa.v, false, kb.v, (short)0, sc, false, false);
  }

  // spill scaled scores to LDS in (row, col) order
  const float scale = 0.125f;  // 1/sqrt(64)
#pragma unroll
  for (int r = 0; r < 8; ++r)
    Sf[h][(r + 8 * hi) * 16 + lrow] = sc[r] * scale;

  // stage V transposed: vT[dh][key]; each lane loads half a V row (32 elems)
  {
    const bf16_t* vp =
        qkv + ((size_t)lrow * S_ + s) * NQKV + 2 * E_ + h * DH_ + hi * 32;
    union { uint4 q[4]; bf16_t e[32]; } vld;
    vld.q[0] = *(const uint4*)(vp);
    vld.q[1] = *(const uint4*)(vp + 8);
    vld.q[2] = *(const uint4*)(vp + 16);
    vld.q[3] = *(const uint4*)(vp + 24);
#pragma unroll
    for (int j = 0; j < 32; ++j)
      vT[h][(hi * 32 + j) * 16 + lrow] = vld.e[j];
  }

  // ---- per-row softmax, one row per lane (lanes 0..15) ----
  if (lane < 16) {
    const float* sr = &Sf[h][lane * 16];
    float v[16];
    *(float4*)(v)      = *(const float4*)(sr);
    *(float4*)(v + 4)  = *(const float4*)(sr + 4);
    *(float4*)(v + 8)  = *(const float4*)(sr + 8);
    *(float4*)(v + 12) = *(const float4*)(sr + 12);
    float mx = v[0];
#pragma unroll
    for (int i = 1; i < 16; ++i) mx = fmaxf(mx, v[i]);
    float sum = 0.f;
#pragma unroll
    for (int i = 0; i < 16; ++i) { v[i] = __expf(v[i] - mx); sum += v[i]; }
    float inv = 1.0f / sum;
    uint4 w0, w1;
    w0.x = pack2bf(v[0] * inv,  v[1] * inv);
    w0.y = pack2bf(v[2] * inv,  v[3] * inv);
    w0.z = pack2bf(v[4] * inv,  v[5] * inv);
    w0.w = pack2bf(v[6] * inv,  v[7] * inv);
    w1.x = pack2bf(v[8] * inv,  v[9] * inv);
    w1.y = pack2bf(v[10] * inv, v[11] * inv);
    w1.z = pack2bf(v[12] * inv, v[13] * inv);
    w1.w = pack2bf(v[14] * inv, v[15] * inv);
    *(uint4*)&Pl[h][lane * 16]     = w0;
    *(uint4*)&Pl[h][lane * 16 + 8] = w1;
  }
  __syncthreads();   // cheap; guarantees Pl/vT visible before fragment loads

  // ---- O = P(16x16, K padded to 32) x V(16x64): 4 WMMAs ----
  Frag pa;
  pa.q[0] = *(const uint4*)&Pl[h][lrow * 16 + khalf];
  pa.q[1] = (uint4){0, 0, 0, 0};                 // zero-pad K 16..31
  v8f oacc[4];
#pragma unroll
  for (int d = 0; d < 4; ++d) {
    Frag vb;
    vb.q[0] = *(const uint4*)&vT[h][(d * 16 + lrow) * 16 + khalf];
    vb.q[1] = (uint4){0, 0, 0, 0};
    v8f z = (v8f){0,0,0,0,0,0,0,0};
    oacc[d] = __builtin_amdgcn_wmma_f32_16x16x32_bf16(
        false, pa.v, false, vb.v, (short)0, z, false, false);
  }

  // ---- store O rows back to per-token layout ----
#pragma unroll
  for (int d = 0; d < 4; ++d)
#pragma unroll
    for (int r = 0; r < 8; ++r) {
      int a  = r + 8 * hi;             // query/batch index
      int dh = d * 16 + lrow;
      o[((size_t)a * S_ + s) * E_ + h * DH_ + dh] = f2bf(oacc[d][r]);
    }
}

// ---------------------------------------------------------------------------
// Host launcher
// ---------------------------------------------------------------------------
extern "C" void kernel_launch(void* const* d_in, const int* in_sizes, int n_in,
                              void* d_out, int out_size, void* d_ws,
                              size_t ws_size, hipStream_t stream) {
  (void)in_sizes; (void)n_in; (void)out_size; (void)ws_size;
  const float* x     = (const float*)d_in[0];
  const float* W_in  = (const float*)d_in[1];
  const float* b_in  = (const float*)d_in[2];
  const float* W_out = (const float*)d_in[3];
  const float* b_out = (const float*)d_in[4];
  float* out = (float*)d_out;

  char* ws = (char*)d_ws;
  bf16_t* Winbf  = (bf16_t*)ws; ws += (size_t)NQKV * E_   * 2;
  bf16_t* Woutbf = (bf16_t*)ws; ws += (size_t)E_   * E_   * 2;
  bf16_t* qkvbf  = (bf16_t*)ws; ws += (size_t)NTOK * NQKV * 2;
  bf16_t* obf    = (bf16_t*)ws;

  cvt_f32_bf16_kernel<<<512, 256, 0, stream>>>(W_in, Winbf, NQKV * E_);
  cvt_f32_bf16_kernel<<<256, 256, 0, stream>>>(W_out, Woutbf, E_ * E_);

  // QKV projection: x (f32, converted in-kernel) x W_in^T -> qkv bf16
  gemm_wmma_kernel<true, true><<<dim3(NTOK / 64, NQKV / 512), 256, 0, stream>>>(
      (const void*)x, Winbf, b_in, (void*)qkvbf, NTOK, NQKV, E_);

  attn_wmma_kernel<<<S_, 256, 0, stream>>>(qkvbf, obf);

  // Output projection: o bf16 x W_out^T -> out f32
  gemm_wmma_kernel<false, false><<<dim3(NTOK / 64, E_ / 512), 256, 0, stream>>>(
      (const void*)obf, Woutbf, b_out, (void*)out, NTOK, E_, E_);
}
